// AttentionSE3_89928025244027
// MI455X (gfx1250) — compile-verified, matrix operand order checked by reading, not run on hardware
//
#include <hip/hip_runtime.h>

// ---------------------------------------------------------------------------
// SE(3) attention layer, fused for MI455X (gfx1250, wave32, WMMA bf16).
// Dominant cost: per-edge radial MLP layer-3 GEMMs (M=16384 edges, K=128,
// N=24576 total cols, ~103 GFLOP) -> v_wmma_f32_16x16x32_bf16, fused with the
// basis contraction so R is never materialized.  B matrices are pre-swizzled
// into fragment-major layout (one 32B vector load per lane per fragment), and
// k3 uses a 32-edge M-tile so each B fragment feeds two WMMAs (halves the
// L2-resident w3 streaming traffic to ~3 GB vs the 16-edge version).
// ---------------------------------------------------------------------------

#define NB    512
#define JN    32
#define EEDGE (NB * JN)   // 16384 edges
#define MIDC  128
#define HIDC  128
#define DIMC  16

typedef __attribute__((ext_vector_type(16))) __bf16 bf16x16;
typedef __attribute__((ext_vector_type(8)))  float  v8f;

__device__ __forceinline__ __bf16 f2bf(float f) {
  union { unsigned short u; __bf16 b; } cvt;
  unsigned u = __float_as_uint(f);
  cvt.u = (unsigned short)((u + 0x7fffu + ((u >> 16) & 1u)) >> 16);  // RNE
  return cvt.b;
}

// ISA 7.12.2 16-bit fragment K index for (element j, lane): two bf16 per VGPR.
__device__ __forceinline__ int frag_k(int j, int half) {
  const int r = j >> 1, c = j & 1;
  return ((r & 3) << 1) + c + ((r >> 2) << 4) + (half << 3);   // 0..31
}

// ---------------------------------------------------------------------------
// Fragment-swizzling f32 -> bf16 conversion for a row-major [128 x ncols] B
// matrix.  Output: for frag = (ct*4 + kc)*32 + lane, 16 contiguous bf16:
//   dst[frag*16 + j] = src[(kc*32 + frag_k(j,lane>>4)) * ncols + ct*16 + (lane&15)]
// ---------------------------------------------------------------------------
__global__ void kswz(const float* __restrict__ src, __bf16* __restrict__ dst,
                     int ncols) {
  const int idx = blockIdx.x * blockDim.x + threadIdx.x;   // fragment-lane id
  const int nfrag = (ncols >> 4) * 4 * 32;
  if (idx >= nfrag) return;
  const int lane = idx & 31;
  const int kc   = (idx >> 5) & 3;
  const int ct   = idx >> 7;
  const int half = lane >> 4, nc = lane & 15;
  const int kb   = kc * 32;
  __bf16* d = dst + (size_t)idx * 16;
#pragma unroll
  for (int j = 0; j < 16; ++j)
    d[j] = f2bf(src[(size_t)(kb + frag_k(j, half)) * ncols + ct * 16 + nc]);
}

__global__ void kzero(float* __restrict__ p, int n) {
  for (int i = blockIdx.x * blockDim.x + threadIdx.x; i < n; i += gridDim.x * blockDim.x)
    p[i] = 0.f;
}

// ---------------------------------------------------------------------------
// Per-edge basis/feature pre-contraction:
//   T[tag][p,i,f] = sum_q xn[i,q] * basis[p,q,f]   (xn gathered via neighbors)
// ---------------------------------------------------------------------------
__global__ void kT_kernel(const float* __restrict__ x0, const float* __restrict__ x1,
                          const int* __restrict__ nb,
                          const float* __restrict__ bas00, const float* __restrict__ bas01,
                          const float* __restrict__ bas10, const float* __restrict__ bas11,
                          float* __restrict__ T00, float* __restrict__ T01,
                          float* __restrict__ T10, float* __restrict__ T11) {
  const int idx = blockIdx.x * blockDim.x + threadIdx.x;
  if (idx >= EEDGE * 16) return;
  const int e = idx >> 4, i = idx & 15;
  const int n = nb[e];
  const float x0v = x0[n * DIMC + i];
  float x1v[3];
#pragma unroll
  for (int q = 0; q < 3; ++q) x1v[q] = x1[(n * DIMC + i) * 3 + q];

  T00[e * 16 + i] = bas00[e] * x0v;                       // [e][i]
  float t10 = 0.f;
#pragma unroll
  for (int q = 0; q < 3; ++q) t10 += x1v[q] * bas10[e * 3 + q];
  T10[e * 16 + i] = t10;                                  // [e][i]
#pragma unroll
  for (int p = 0; p < 3; ++p)
    T01[(e * 3 + p) * 16 + i] = bas01[e * 3 + p] * x0v;   // [e][p][i]
#pragma unroll
  for (int p = 0; p < 3; ++p) {
#pragma unroll
    for (int f = 0; f < 3; ++f) {
      float s = 0.f;
#pragma unroll
      for (int q = 0; q < 3; ++q) s += x1v[q] * bas11[((e * 3 + p) * 3 + q) * 3 + f];
      T11[((e * 3 + p) * 16 + i) * 3 + f] = s;            // [e][p][i][f]
    }
  }
}

// ---------------------------------------------------------------------------
// Radial MLP layers 1+2, fused.  grid=(1024 edge-tiles, 8 radial funcs),
// block=256 (8 waves).  Layer1 recomputed in LDS, layer2 = WMMA bf16 GEMM.
// ---------------------------------------------------------------------------
__global__ __launch_bounds__(256) void k2_radial(
    const float* __restrict__ rel_dist,
    const float* __restrict__ w1, const float* __restrict__ b1,
    const float* __restrict__ g1, const float* __restrict__ be1,
    const __bf16* __restrict__ w2s, const float* __restrict__ b2,
    const float* __restrict__ g2, const float* __restrict__ be2,
    __bf16* __restrict__ h2out) {
  const int tile = blockIdx.x;
  const int k    = blockIdx.y;
  const int tid  = threadIdx.x;
  const int wave = tid >> 5, lane = tid & 31;
  const int half = lane >> 4, mr = lane & 15, nc = lane & 15;

  __shared__ __bf16 h1[16][MIDC + 8];   // padded vs LDS bank conflicts
  __shared__ float  cbuf[16][MIDC];

  // ---- layer 1: h1 = relu(LN(feat*w1 + b1)); each wave owns 2 edges
  for (int sub = 0; sub < 2; ++sub) {
    const int el = wave * 2 + sub;
    const float feat = rel_dist[tile * 16 + el];
    float v[4]; float s = 0.f, s2 = 0.f;
#pragma unroll
    for (int t = 0; t < 4; ++t) {
      const int m = lane + 32 * t;
      const float x = feat * w1[k * MIDC + m] + b1[k * MIDC + m];
      v[t] = x; s += x; s2 += x * x;
    }
#pragma unroll
    for (int d = 16; d >= 1; d >>= 1) { s += __shfl_xor(s, d); s2 += __shfl_xor(s2, d); }
    const float mu = s * (1.f / MIDC);
    const float var = s2 * (1.f / MIDC) - mu * mu;
    const float rstd = rsqrtf(var + 1e-5f);
#pragma unroll
    for (int t = 0; t < 4; ++t) {
      const int m = lane + 32 * t;
      const float y = (v[t] - mu) * rstd * g1[k * MIDC + m] + be1[k * MIDC + m];
      h1[el][m] = f2bf(fmaxf(y, 0.f));
    }
  }
  __syncthreads();

  // ---- layer 2 GEMM: wave w -> col tile w; B fragments pre-swizzled
  const bf16x16* Wf = reinterpret_cast<const bf16x16*>(w2s) + ((size_t)k * 8 + wave) * 4 * 32;
  v8f acc = {};
  const int col0 = wave * 16;
#pragma unroll
  for (int kc = 0; kc < 4; ++kc) {
    bf16x16 a;
#pragma unroll
    for (int j = 0; j < 16; ++j) a[j] = h1[mr][kc * 32 + frag_k(j, half)];
    const bf16x16 bm = Wf[kc * 32 + lane];
    acc = __builtin_amdgcn_wmma_f32_16x16x32_bf16(false, a, false, bm, (short)0, acc,
                                                  false, false);
  }
#pragma unroll
  for (int j = 0; j < 8; ++j) {
    const int row = j + (half << 3);   // C/D: VGPR j -> row j (+8 for hi half)
    cbuf[row][col0 + nc] = acc[j] + b2[k * MIDC + col0 + nc];
  }
  __syncthreads();

  // ---- LN + ReLU -> h2 (bf16)
  for (int sub = 0; sub < 2; ++sub) {
    const int el = wave * 2 + sub;
    float v[4]; float s = 0.f, s2 = 0.f;
#pragma unroll
    for (int t = 0; t < 4; ++t) { const int m = lane + 32 * t; const float x = cbuf[el][m];
                                  v[t] = x; s += x; s2 += x * x; }
#pragma unroll
    for (int d = 16; d >= 1; d >>= 1) { s += __shfl_xor(s, d); s2 += __shfl_xor(s2, d); }
    const float mu = s * (1.f / MIDC), var = s2 * (1.f / MIDC) - mu * mu;
    const float rstd = rsqrtf(var + 1e-5f);
#pragma unroll
    for (int t = 0; t < 4; ++t) {
      const int m = lane + 32 * t;
      const float y = (v[t] - mu) * rstd * g2[k * MIDC + m] + be2[k * MIDC + m];
      h2out[((size_t)k * EEDGE + tile * 16 + el) * MIDC + m] = f2bf(fmaxf(y, 0.f));
    }
  }
}

// ---------------------------------------------------------------------------
// Big fused kernel: R = h2 @ w3[k] + b3 via WMMA tiles, contracted on the fly
// against T -> k0/v0/k1/v1.  grid=(512 edge-tiles of 32, 8), block=256.
// Each wave holds TWO 16-row A-fragment sets; every B fragment feeds 2 WMMAs.
// ---------------------------------------------------------------------------
__global__ __launch_bounds__(256) void k3_conv(
    const __bf16* __restrict__ h2, const __bf16* __restrict__ w3s,
    const float* b3_0, const float* b3_1, const float* b3_2, const float* b3_3,
    const float* b3_4, const float* b3_5, const float* b3_6, const float* b3_7,
    const float* __restrict__ T00, const float* __restrict__ T01,
    const float* __restrict__ T10, const float* __restrict__ T11,
    float* __restrict__ k0b, float* __restrict__ k1b,
    float* __restrict__ v0b, float* __restrict__ v1b) {
  const int tile = blockIdx.x, kk = blockIdx.y;   // tile of 32 edges
  const int tag = kk & 3;
  const int P = (tag == 1 || tag == 3) ? 3 : 1;   // output m-dim
  const int F = (tag == 3) ? 3 : 1;               // frequency dim
  const int tid = threadIdx.x, wave = tid >> 5, lane = tid & 31;
  const int half = lane >> 4, mr = lane & 15, nc = lane & 15;
  const int e0 = tile * 32;

  __shared__ __bf16 ha[32][MIDC + 8];
  __shared__ float  Ts[32][3][16][3];
  __shared__ float  outacc[32][HIDC][3];

  // stage h2 tile (32 edges x 128) as 4-bf16 (uint2) vector copies
  for (int idx = tid; idx < 32 * (MIDC / 4); idx += 256) {
    const int el = idx >> 5, m4 = idx & 31;
    *(uint2*)&ha[el][m4 * 4] =
        *(const uint2*)&h2[((size_t)kk * EEDGE + e0 + el) * MIDC + m4 * 4];
  }
  for (int idx = tid; idx < 32 * HIDC * 3; idx += 256) (&outacc[0][0][0])[idx] = 0.f;
  if (tag == 0 || tag == 2) {
    const float* T = (tag == 0) ? T00 : T10;
    for (int idx = tid; idx < 32 * 16; idx += 256) {
      const int el = idx >> 4, i = idx & 15;
      Ts[el][0][i][0] = T[(e0 + el) * 16 + i];
    }
  } else if (tag == 1) {
    for (int idx = tid; idx < 32 * 48; idx += 256) {
      const int el = idx / 48, r = idx % 48;
      Ts[el][r / 16][r % 16][0] = T01[(e0 + el) * 48 + r];
    }
  } else {
    for (int idx = tid; idx < 32 * 144; idx += 256) {
      const int el = idx / 144, r = idx % 144;
      Ts[el][r / 48][(r % 48) / 3][r % 3] = T11[(e0 + el) * 144 + r];
    }
  }
  __syncthreads();

  const int ncols = F * 2048;
  const bf16x16* Wf = reinterpret_cast<const bf16x16*>(
      w3s + (size_t)(2048 * kk + ((kk >= 4) ? 4096 : 0)) * MIDC);
  const float* b3tab[8] = {b3_0, b3_1, b3_2, b3_3, b3_4, b3_5, b3_6, b3_7};
  const float* b3 = b3tab[kk];
  const int ntiles = ncols >> 4;                                // 128 or 384

  // preload A fragments for both 16-row subtiles (reused for every col tile)
  bf16x16 afr[2][4];
#pragma unroll
  for (int s = 0; s < 2; ++s)
#pragma unroll
    for (int kc = 0; kc < 4; ++kc)
#pragma unroll
      for (int j = 0; j < 16; ++j)
        afr[s][kc][j] = ha[s * 16 + mr][kc * 32 + frag_k(j, half)];

  for (int ct = wave; ct < ntiles; ct += 8) {
    v8f acc[2]; acc[0] = (v8f){}; acc[1] = (v8f){};
#pragma unroll
    for (int kc = 0; kc < 4; ++kc) {
      const bf16x16 bm = Wf[((size_t)ct * 4 + kc) * 32 + lane];   // one 32B load
      acc[0] = __builtin_amdgcn_wmma_f32_16x16x32_bf16(false, afr[0][kc], false, bm,
                                                       (short)0, acc[0], false, false);
      acc[1] = __builtin_amdgcn_wmma_f32_16x16x32_bf16(false, afr[1][kc], false, bm,
                                                       (short)0, acc[1], false, false);
    }
    // ---- contraction: column c -> (o,i,fi); c = (o*16+i)*F + fi
    const int c = (ct << 4) + nc;
    int o, i, fi;
    if (F == 1) { o = c >> 4; i = c & 15; fi = 0; }
    else        { o = c / 48; const int rem = c % 48; i = rem / 3; fi = rem % 3; }
    const float bias = b3[c];
#pragma unroll
    for (int s = 0; s < 2; ++s) {
#pragma unroll
      for (int j = 0; j < 8; ++j) {
        const int el = s * 16 + j + (half << 3);
        const float val = acc[s][j] + bias;         // R[e,o,i,fi]
        for (int p = 0; p < P; ++p) {               // P uniform across block
          float contrib = val * Ts[el][p][i][fi];
#pragma unroll
          for (int d = 8; d >= 1; d >>= 1) contrib += __shfl_xor(contrib, d);
          if (nc == 0) atomicAdd(&outacc[el][o][p], contrib);
        }
      }
    }
  }
  __syncthreads();

  // tags 00+10 -> deg0, 01+11 -> deg1; accumulate across tags via global atomics
  float* dst = (tag == 0 || tag == 2) ? ((kk < 4) ? k0b : v0b)
                                      : ((kk < 4) ? k1b : v1b);
  for (int idx = tid; idx < 32 * HIDC * P; idx += 256) {
    const int el = idx / (HIDC * P), r = idx % (HIDC * P);
    const int o = r / P, p = r % P;
    atomicAdd(&dst[((size_t)(e0 + el) * HIDC + o) * P + p], outacc[el][o][p]);
  }
}

// ---------------------------------------------------------------------------
// Attention + output projection.  block=512 (16 waves): wave = (deg, head),
// lane = neighbor j.  Masked softmax over 32 lanes via shuffles.
// ---------------------------------------------------------------------------
__global__ __launch_bounds__(512) void k4_attn(
    const float* __restrict__ x0, const float* __restrict__ x1,
    const float* __restrict__ wq0, const float* __restrict__ wq1,
    const float* __restrict__ k0b, const float* __restrict__ k1b,
    const float* __restrict__ v0b, const float* __restrict__ v1b,
    const unsigned char* __restrict__ mask,
    const float* __restrict__ wout0, const float* __restrict__ wout1,
    float* __restrict__ out) {
  const int n = blockIdx.x;
  const int tid = threadIdx.x, wave = tid >> 5, lane = tid & 31;
  __shared__ float q0[HIDC], q1[HIDC][3], o0[HIDC], o1[HIDC][3];

  for (int idx = tid; idx < HIDC; idx += 512) {
    float s = 0.f;
#pragma unroll
    for (int d = 0; d < DIMC; ++d) s += x0[n * DIMC + d] * wq0[d * HIDC + idx];
    q0[idx] = s;
  }
  for (int idx = tid; idx < HIDC * 3; idx += 512) {
    const int o = idx / 3, m = idx % 3;
    float s = 0.f;
#pragma unroll
    for (int d = 0; d < DIMC; ++d) s += x1[(n * DIMC + d) * 3 + m] * wq1[d * HIDC + o];
    q1[o][m] = s;
  }
  __syncthreads();

  const int deg = wave >> 3, h = wave & 7;
  const int e = n * JN + lane;
  float sim = 0.f;
  if (deg == 0) {
#pragma unroll
    for (int d = 0; d < DIMC; ++d) {
      const int o = h * DIMC + d;
      sim += q0[o] * k0b[(size_t)e * HIDC + o];
    }
  } else {
#pragma unroll
    for (int d = 0; d < DIMC; ++d) {
      const int o = h * DIMC + d;
#pragma unroll
      for (int m = 0; m < 3; ++m) sim += q1[o][m] * k1b[((size_t)e * HIDC + o) * 3 + m];
    }
  }
  sim *= 0.25f;                              // DHEAD^-0.5 = 1/4
  if (!mask[e]) sim = -3.402823466e38f;
  float mx = sim;
#pragma unroll
  for (int d = 16; d >= 1; d >>= 1) mx = fmaxf(mx, __shfl_xor(mx, d));
  const float ex = __expf(sim - mx);
  float den = ex;
#pragma unroll
  for (int d = 16; d >= 1; d >>= 1) den += __shfl_xor(den, d);
  const float w = ex / den;

  if (deg == 0) {
#pragma unroll
    for (int d = 0; d < DIMC; ++d) {
      const int o = h * DIMC + d;
      float acc = w * v0b[(size_t)e * HIDC + o];
#pragma unroll
      for (int dd = 16; dd >= 1; dd >>= 1) acc += __shfl_xor(acc, dd);
      if (lane == 0) o0[o] = acc;
    }
  } else {
#pragma unroll
    for (int d = 0; d < DIMC; ++d) {
      const int o = h * DIMC + d;
#pragma unroll
      for (int m = 0; m < 3; ++m) {
        float acc = w * v1b[((size_t)e * HIDC + o) * 3 + m];
#pragma unroll
        for (int dd = 16; dd >= 1; dd >>= 1) acc += __shfl_xor(acc, dd);
        if (lane == 0) o1[o][m] = acc;
      }
    }
  }
  __syncthreads();

  for (int idx = tid; idx < DIMC; idx += 512) {
    float s = 0.f;
    for (int o = 0; o < HIDC; ++o) s += o0[o] * wout0[o * DIMC + idx];
    out[n * DIMC + idx] = s;
  }
  for (int idx = tid; idx < DIMC * 3; idx += 512) {
    const int d = idx / 3, m = idx % 3;
    float s = 0.f;
    for (int o = 0; o < HIDC; ++o) s += o1[o][m] * wout1[o * DIMC + d];
    out[NB * DIMC + (n * DIMC + d) * 3 + m] = s;
  }
}

// ---------------------------------------------------------------------------
// Host launcher.  Workspace layout (bytes, all 256-aligned):
//   w2s 262144 | w3s 6291456 | h2 33554432 | T00 1M | T10 1M | T01 3M |
//   T11 9.4M | k0 8.4M | v0 8.4M | k1 25.2M | v1 25.2M   (~116 MB total)
// ---------------------------------------------------------------------------
extern "C" void kernel_launch(void* const* d_in, const int* in_sizes, int n_in,
                              void* d_out, int out_size, void* d_ws, size_t ws_size,
                              hipStream_t stream) {
  (void)in_sizes; (void)n_in; (void)out_size; (void)ws_size;

  const float* x0      = (const float*)d_in[0];
  const float* x1      = (const float*)d_in[1];
  const float* rel     = (const float*)d_in[2];
  const float* bas00   = (const float*)d_in[3];
  const float* bas01   = (const float*)d_in[4];
  const float* bas10   = (const float*)d_in[5];
  const float* bas11   = (const float*)d_in[6];
  const float* rf_w1   = (const float*)d_in[7];
  const float* rf_b1   = (const float*)d_in[8];
  const float* rf_g1   = (const float*)d_in[9];
  const float* rf_be1  = (const float*)d_in[10];
  const float* rf_w2   = (const float*)d_in[11];
  const float* rf_b2   = (const float*)d_in[12];
  const float* rf_g2   = (const float*)d_in[13];
  const float* rf_be2  = (const float*)d_in[14];
  // w3/b3 pairs: indices 15..30 in order k00,k01,k10,k11,v00,v01,v10,v11
  const float* wq0     = (const float*)d_in[31];
  const float* wq1     = (const float*)d_in[32];
  const float* wout0   = (const float*)d_in[33];
  const float* wout1   = (const float*)d_in[34];
  const int*   nb      = (const int*)d_in[35];
  const unsigned char* msk = (const unsigned char*)d_in[36];  // jax bool = 1B

  char* ws = (char*)d_ws;
  size_t off = 0;
  __bf16* w2s = (__bf16*)(ws + off); off += 262144;
  __bf16* w3s = (__bf16*)(ws + off); off += 6291456;
  __bf16* h2  = (__bf16*)(ws + off); off += 33554432;
  float* T00 = (float*)(ws + off); off += 1048576;
  float* T10 = (float*)(ws + off); off += 1048576;
  float* T01 = (float*)(ws + off); off += 3145728;
  float* T11 = (float*)(ws + off); off += 9437184;
  float* k0b = (float*)(ws + off); off += 8388608;
  float* v0b = (float*)(ws + off); off += 8388608;
  float* k1b = (float*)(ws + off); off += 25165824;
  float* v1b = (float*)(ws + off); off += 25165824;

  // weight conversion + fragment swizzle (w2: 8 x [128x128], w3: 8 matrices)
  for (int k = 0; k < 8; ++k) {
    const int nfrag = (128 >> 4) * 4 * 32;   // 1024
    kswz<<<(nfrag + 255) / 256, 256, 0, stream>>>(rf_w2 + (size_t)k * 128 * 128,
                                                  w2s + (size_t)k * 16384, 128);
  }
  for (int mk = 0; mk < 8; ++mk) {
    const int cols = ((mk & 3) == 3) ? 6144 : 2048;
    const int nfrag = (cols >> 4) * 4 * 32;
    const size_t coff = (size_t)(2048 * mk + ((mk >= 4) ? 4096 : 0)) * 128;
    kswz<<<(nfrag + 255) / 256, 256, 0, stream>>>((const float*)d_in[15 + 2 * mk],
                                                  w3s + coff, cols);
  }
  // zero k/v accumulators (contiguous region: 16 777 216 floats)
  kzero<<<4096, 256, 0, stream>>>(k0b, 16777216);

  // basis/feature pre-contraction
  kT_kernel<<<(EEDGE * 16 + 255) / 256, 256, 0, stream>>>(
      x0, x1, nb, bas00, bas01, bas10, bas11, T00, T01, T10, T11);

  // radial MLP (layers 1+2, WMMA)
  dim3 g2(EEDGE / 16, 8);
  k2_radial<<<g2, 256, 0, stream>>>(rel, rf_w1, rf_b1, rf_g1, rf_be1,
                                    w2s, rf_b2, rf_g2, rf_be2, h2);

  // layer-3 GEMM + conv contraction (WMMA, fused, 32-edge M-tile)
  dim3 g3(EEDGE / 32, 8);
  k3_conv<<<g3, 256, 0, stream>>>(
      h2, w3s,
      (const float*)d_in[16], (const float*)d_in[18], (const float*)d_in[20],
      (const float*)d_in[22], (const float*)d_in[24], (const float*)d_in[26],
      (const float*)d_in[28], (const float*)d_in[30],
      T00, T01, T10, T11, k0b, k1b, v0b, v1b);

  // attention + output projections
  k4_attn<<<NB, 512, 0, stream>>>(x0, x1, wq0, wq1, k0b, k1b, v0b, v1b,
                                  msk, wout0, wout1, (float*)d_out);
}
